// ScaledDotProductAttention_30073361007055
// MI455X (gfx1250) — matrix operand order repaired and checked by reading, not compiled, not run on hardware
//
#include <hip/hip_runtime.h>

// ---------------------------------------------------------------------------
// ScaledDotProductAttention for MI455X (gfx1250, wave32, WMMA bf16)
//   B=4, T=64, L=4096, D=4096
//   Pass 1: scores = Q @ mask^T * (1/64), masked          -> ws (fp32, 4MB)
//   Pass 2: row softmax                                    -> ws (bf16, 2MB)
//   Pass 3: context = attn @ mask                          -> d_out (fp32)
// ---------------------------------------------------------------------------

typedef __attribute__((ext_vector_type(16))) __bf16 v16bf;
typedef __attribute__((ext_vector_type(8)))  __bf16 v8bf;
typedef __attribute__((ext_vector_type(4)))  __bf16 v4bf;
typedef __attribute__((ext_vector_type(8)))  float  v8f;
typedef __attribute__((ext_vector_type(4)))  float  v4f;

#define BATCH 4
#define TT    64
#define LL    4096
#define DD    4096
#define MASK_VALUE (-1.0e9f)

__device__ __forceinline__ v8bf cat4(v4bf a, v4bf b) {
  return __builtin_shufflevector(a, b, 0, 1, 2, 3, 4, 5, 6, 7);
}
__device__ __forceinline__ v16bf cat8(v8bf lo, v8bf hi) {
  return __builtin_shufflevector(lo, hi, 0, 1, 2, 3, 4, 5, 6, 7,
                                         8, 9, 10, 11, 12, 13, 14, 15);
}

// Native fp32 -> bf16 (lets the backend use packed v_cvt_*bf16* ops).
__device__ __forceinline__ v16bf cvt16(v4f a, v4f b, v4f c, v4f d) {
  return cat8(cat4(__builtin_convertvector(a, v4bf),
                   __builtin_convertvector(b, v4bf)),
              cat4(__builtin_convertvector(c, v4bf),
                   __builtin_convertvector(d, v4bf)));
}
__device__ __forceinline__ unsigned short bf16_bits(float f) {
  return __builtin_bit_cast(unsigned short, (__bf16)f);
}

__device__ __forceinline__ v8f wmma_bf16(v16bf a, v16bf b, v8f c) {
  // (neg_a, A, neg_b, B, c_mod, C, reuse_a, reuse_b)
  return __builtin_amdgcn_wmma_f32_16x16x32_bf16(false, a, false, b,
                                                 (short)0, c, false, false);
}

// ---------------------------------------------------------------------------
// Pass 1: scores[b, t, l] = (Q[t,:] . mask[b,l,:]) / 64, masked.
// Block = 128 threads (4 waves). Block tile: T=64 x L=32.
// Wave w: rows [16w, 16w+16), two 16-wide column tiles -> 2 WMMA accums.
// Fragments are loaded from global in a contiguous k order, same for A and B:
//   A 16x32 bf16: lane m,              halves j    -> K=16*hi+j (contiguous)
//   B 32x16 bf16: lane n = column,     halves j    -> K=16*hi+j (contiguous)
// ---------------------------------------------------------------------------
__global__ __launch_bounds__(128) void scores_kernel(
    const float* __restrict__ mask, const float* __restrict__ Q,
    const unsigned char* __restrict__ am, float* __restrict__ scores) {
  const int b    = blockIdx.y;
  const int l0   = blockIdx.x * 32;
  const int w    = threadIdx.x >> 5;
  const int lane = threadIdx.x & 31;
  const int lo   = lane & 15;
  const int hi   = lane >> 4;

  const float* qrow  = Q + (size_t)(w * 16 + lo) * DD;
  const float* krow0 = mask + (size_t)b * LL * DD + (size_t)(l0 + lo) * DD;
  const float* krow1 = krow0 + (size_t)16 * DD;

  v8f acc0 = {};
  v8f acc1 = {};

  for (int k0 = 0; k0 < DD; k0 += 32) {
    const float* qa = qrow + k0 + 16 * hi;
    v16bf afrag = cvt16(*(const v4f*)(qa), *(const v4f*)(qa + 4),
                        *(const v4f*)(qa + 8), *(const v4f*)(qa + 12));

    const float* p0 = krow0 + k0 + 16 * hi;
    v16bf bfrag0 = cvt16(*(const v4f*)(p0), *(const v4f*)(p0 + 4),
                         *(const v4f*)(p0 + 8), *(const v4f*)(p0 + 12));
    const float* p1 = krow1 + k0 + 16 * hi;
    v16bf bfrag1 = cvt16(*(const v4f*)(p1), *(const v4f*)(p1 + 4),
                         *(const v4f*)(p1 + 8), *(const v4f*)(p1 + 12));

    // Stream-ahead prefetch of the mask rows (global_prefetch_b8).
    __builtin_prefetch(p0 + 256, 0, 0);
    __builtin_prefetch(p1 + 256, 0, 0);

    acc0 = wmma_bf16(afrag, bfrag0, acc0);
    acc1 = wmma_bf16(afrag, bfrag1, acc1);
  }

  const float scale = 1.0f / 64.0f;  // 1/sqrt(4096)
#pragma unroll
  for (int r = 0; r < 8; ++r) {
    const int t = w * 16 + r + 8 * hi;  // C layout: lanes 16-31 hold M=r+8
    const size_t row = (size_t)(b * TT + t) * LL;
    size_t i0 = row + l0 + lo;
    size_t i1 = row + l0 + 16 + lo;
    float s0 = acc0[r] * scale;
    float s1 = acc1[r] * scale;
    if (am[i0]) s0 = MASK_VALUE;
    if (am[i1]) s1 = MASK_VALUE;
    scores[i0] = s0;
    scores[i1] = s1;
  }
}

// ---------------------------------------------------------------------------
// Pass 2: row softmax over L=4096. One block (256 thr) per row (B*T = 256).
// Writes normalized probabilities as bf16 (pass-3 A operand format).
// ---------------------------------------------------------------------------
__global__ __launch_bounds__(256) void softmax_kernel(
    const float* __restrict__ scores, unsigned short* __restrict__ attn) {
  const int row = blockIdx.x;  // b*64 + t
  const int tid = threadIdx.x;
  const float* src = scores + (size_t)row * LL + tid * 16;

  v4f v[4];
#pragma unroll
  for (int i = 0; i < 4; ++i) v[i] = *(const v4f*)(src + i * 4);

  float m = -3.4e38f;
#pragma unroll
  for (int i = 0; i < 4; ++i)
#pragma unroll
    for (int j = 0; j < 4; ++j) m = fmaxf(m, v[i][j]);

#pragma unroll
  for (int off = 16; off > 0; off >>= 1) m = fmaxf(m, __shfl_xor(m, off, 32));

  __shared__ float red_max[8];
  __shared__ float red_sum[8];
  if ((tid & 31) == 0) red_max[tid >> 5] = m;
  __syncthreads();
  float mrow = red_max[0];
#pragma unroll
  for (int i = 1; i < 8; ++i) mrow = fmaxf(mrow, red_max[i]);

  float e[16];
  float s = 0.0f;
#pragma unroll
  for (int i = 0; i < 4; ++i)
#pragma unroll
    for (int j = 0; j < 4; ++j) {
      float ev = __expf(v[i][j] - mrow);
      e[i * 4 + j] = ev;
      s += ev;
    }
#pragma unroll
  for (int off = 16; off > 0; off >>= 1) s += __shfl_xor(s, off, 32);
  if ((tid & 31) == 0) red_sum[tid >> 5] = s;
  __syncthreads();
  float total = 0.0f;
#pragma unroll
  for (int i = 0; i < 8; ++i) total += red_sum[i];
  const float inv = 1.0f / total;

  unsigned* dst = (unsigned*)(attn + (size_t)row * LL + tid * 16);
#pragma unroll
  for (int i = 0; i < 8; ++i) {
    unsigned u0 = bf16_bits(e[2 * i] * inv);
    unsigned u1 = bf16_bits(e[2 * i + 1] * inv);
    dst[i] = u0 | (u1 << 16);
  }
}

// ---------------------------------------------------------------------------
// Pass 3: context[b, t, d] = sum_l attn[b,t,l] * mask[b,l,d].
// Block = 128 threads (4 waves). Block tile: T=64 x D=32, K-loop over L.
// mask (row-major K x N) is staged transposed as bf16 into a double-buffered
// LDS tile -> B fragments are two aligned ds_load_b128; one barrier per step.
// ---------------------------------------------------------------------------
__device__ __forceinline__ void store_tile(unsigned short* t, v4f m0, v4f m1,
                                           int lld, int g) {
  v4bf c0 = __builtin_convertvector(m0, v4bf);
  v4bf c1 = __builtin_convertvector(m1, v4bf);
#pragma unroll
  for (int i = 0; i < 4; ++i) {
    t[(8 * g + i) * 32 + lld]     = __builtin_bit_cast(unsigned short, c0[i]);
    t[(8 * g + 4 + i) * 32 + lld] = __builtin_bit_cast(unsigned short, c1[i]);
  }
}

__global__ __launch_bounds__(128) void context_kernel(
    const float* __restrict__ mask, const unsigned short* __restrict__ attn,
    float* __restrict__ out) {
  const int b    = blockIdx.y;
  const int d0   = blockIdx.x * 32;
  const int w    = threadIdx.x >> 5;
  const int lane = threadIdx.x & 31;
  const int lo   = lane & 15;
  const int hi   = lane >> 4;

  __shared__ unsigned short tile[2][32 * 32];  // [d_local][l_local], bf16 bits

  const unsigned short* arow = attn + (size_t)(b * TT + w * 16 + lo) * LL;
  const float* mb = mask + (size_t)b * LL * DD;

  const int lld = threadIdx.x >> 2;  // 0..31 : l_local row to stage
  const int g   = threadIdx.x & 3;   // 0..3  : d cols 8g..8g+7
  const float* srcbase = mb + (size_t)lld * DD + d0 + 8 * g;

  // Prologue: stage tile for k0 = 0 into buffer 0.
  {
    v4f m0 = *(const v4f*)(srcbase);
    v4f m1 = *(const v4f*)(srcbase + 4);
    store_tile(tile[0], m0, m1, lld, g);
  }

  v8f acc0 = {};
  v8f acc1 = {};

  for (int k0 = 0; k0 < LL; k0 += 32) {
    const int cur  = (k0 >> 5) & 1;
    const bool more = (k0 + 32) < LL;

    // Issue next tile's global loads before the barrier (latency overlap).
    v4f n0 = {}, n1 = {};
    if (more) {
      const float* src = srcbase + (size_t)(k0 + 32) * DD;
      n0 = *(const v4f*)(src);
      n1 = *(const v4f*)(src + 4);
      __builtin_prefetch(src + (size_t)32 * DD, 0, 0);
    }

    __syncthreads();  // tile[cur] complete; prior reads of tile[cur^1] done

    // A fragment: attn already bf16 in memory, two b128 loads.
    v8bf alo = *(const v8bf*)(arow + k0 + 16 * hi);
    v8bf ahi = *(const v8bf*)(arow + k0 + 16 * hi + 8);
    v16bf afrag = cat8(alo, ahi);

    // B fragments from transposed LDS tile (aligned ds_load_b128 pairs).
    const unsigned short* bp0 = &tile[cur][lo * 32 + 16 * hi];
    v16bf bfrag0 = cat8(*(const v8bf*)(bp0), *(const v8bf*)(bp0 + 8));
    const unsigned short* bp1 = &tile[cur][(16 + lo) * 32 + 16 * hi];
    v16bf bfrag1 = cat8(*(const v8bf*)(bp1), *(const v8bf*)(bp1 + 8));

    acc0 = wmma_bf16(afrag, bfrag0, acc0);
    acc1 = wmma_bf16(afrag, bfrag1, acc1);

    if (more) store_tile(tile[cur ^ 1], n0, n1, lld, g);
  }

#pragma unroll
  for (int r = 0; r < 8; ++r) {
    const int t = w * 16 + r + 8 * hi;
    const size_t row = (size_t)(b * TT + t) * DD;
    out[row + d0 + lo]      = acc0[r];
    out[row + d0 + 16 + lo] = acc1[r];
  }
}

// ---------------------------------------------------------------------------
extern "C" void kernel_launch(void* const* d_in, const int* in_sizes, int n_in,
                              void* d_out, int out_size, void* d_ws,
                              size_t ws_size, hipStream_t stream) {
  (void)in_sizes; (void)n_in; (void)out_size; (void)ws_size;

  const float* mask       = (const float*)d_in[0];          // [B, L, D] fp32
  const float* Q          = (const float*)d_in[1];          // [1, T, D] fp32
  const unsigned char* am = (const unsigned char*)d_in[2];  // [B, T, L] bool

  float* out = (float*)d_out;  // [B, T, D] fp32

  // Workspace: fp32 scores (4 MB) then bf16 attn (2 MB).
  float* scores = (float*)d_ws;
  unsigned short* attn =
      (unsigned short*)((char*)d_ws + (size_t)BATCH * TT * LL * sizeof(float));

  dim3 g1(LL / 32, BATCH);
  scores_kernel<<<g1, 128, 0, stream>>>(mask, Q, am, scores);

  softmax_kernel<<<BATCH * TT, 256, 0, stream>>>(scores, attn);

  dim3 g3(DD / 32, BATCH);
  context_kernel<<<g3, 128, 0, stream>>>(mask, attn, out);
}
